// GIN_30949534335552
// MI455X (gfx1250) — compile-verified, matrix operand order checked by reading, not run on hardware
//
#include <hip/hip_runtime.h>

// ---------------------------------------------------------------------------
// GIN forward for MI455X (gfx1250).
//  - bf16 WMMA (v_wmma_f32_16x16x32_bf16) for all H=128 GEMMs, f32 accumulate
//  - weights pre-swizzled into exact WMMA B-fragment layout (L2-resident)
//  - GIN combine / mean-div / bias / BN / ReLU fused into GEMM prologue/epilogue
//  - edge scatter-add via native f32 global atomics (h & agg live in 192MB L2)
//  - A-fragment pack is branchless: out-of-range rows are clamped (their D rows
//    are never stored, and WMMA rows are independent), and f32->bf16 uses
//    __builtin_convertvector so the backend emits packed v_cvt_pk_bf16_f32.
// ---------------------------------------------------------------------------

#define HDIM 128
#define NNODES 100000
#define NEDGES 1600000
#define NGRAPH 512
#define BN_EPS_F 1e-5f

typedef __attribute__((ext_vector_type(16))) __bf16 v16bf;
typedef __attribute__((ext_vector_type(16))) float  v16f;
typedef __attribute__((ext_vector_type(8)))  float  v8f;
typedef __attribute__((ext_vector_type(4)))  float  v4f;
typedef __attribute__((ext_vector_type(4)))  unsigned int v4u;

// K offset of VGPR-pair v (0..7) for lane group grp (0: lanes 0-15, 1: 16-31)
// per CDNA5 ISA "16-bit A-Matrix 16x32" layout (B mirrors it by symmetry).
__device__ __forceinline__ int kmap(int vpair, int grp) {
  int k = (vpair < 4) ? (2 * vpair) : (2 * vpair + 8);
  return k + grp * 8;
}

// ---------------------------------------------------------------------------
// Convert the nine 128x128 f32 weight matrices into bf16 fragment order:
// Wfrag[mat][kk(4)][j(8)][lane(32)][16 bf16]  -> each lane's B fragment is a
// contiguous 32B run => two b128 loads, no LDS, perfect L2 reuse.
// ---------------------------------------------------------------------------
__global__ __launch_bounds__(1024) void cvt_weights(
    const float* __restrict__ encW, const float* __restrict__ mlp1W,
    const float* __restrict__ mlp2W, const float* __restrict__ headW1,
    const float* __restrict__ headW2, unsigned short* __restrict__ Wfrag)
{
  const int mi = blockIdx.x;  // 0..8
  const float* W;
  if (mi == 0)      W = encW;
  else if (mi <= 3) W = mlp1W + (size_t)(mi - 1) * HDIM * HDIM;
  else if (mi <= 6) W = mlp2W + (size_t)(mi - 4) * HDIM * HDIM;
  else if (mi == 7) W = headW1;
  else              W = headW2;

  const int t    = threadIdx.x;      // 0..1023 => one (kk,j,lane) triple
  const int kk   = t >> 8;
  const int j    = (t >> 5) & 7;
  const int lane = t & 31;
  const int grp  = lane >> 4;
  const int n    = j * 16 + (lane & 15);

  v16f vf;
#pragma unroll
  for (int e = 0; e < 16; ++e) {
    int k = kk * 32 + kmap(e >> 1, grp) + (e & 1);
    vf[e] = W[(size_t)k * HDIM + n];
  }
  union { v16bf v; v4u q[2]; } o;
  o.v = __builtin_convertvector(vf, v16bf);
  v4u* outp = (v4u*)(Wfrag + (size_t)mi * 16384 +
                     (size_t)((kk * 8 + j) * 32 + lane) * 16);
  outp[0] = o.q[0];
  outp[1] = o.q[1];
}

// ---------------------------------------------------------------------------
// Fused GEMM: out[M,128] = f(A' @ W + bias), A' per COMB mode:
//   COMB 0: A
//   COMB 1: (1+eps)*A + A2           (GIN combine; aux = &eps[layer])
//   COMB 2: A / max(aux[row],1)      (mean pool divide; aux = counts)
// BN 0: bias only.  BN 1: relu(batchnorm(.)) with running stats.
// One 256-thread block = 128 rows; wave w owns rows [w*16, w*16+16) x all 128
// cols => 8 f32 16x16 accumulators, 4 k-steps of bf16 WMMA.
// ---------------------------------------------------------------------------
template <int COMB, int BN>
__global__ __launch_bounds__(256) void gemm128(
    const float* __restrict__ A, const float* __restrict__ A2,
    const float* __restrict__ aux,
    const unsigned short* __restrict__ Wfrag,
    const float* __restrict__ bias,
    const float* __restrict__ bng, const float* __restrict__ bnb,
    const float* __restrict__ bnm, const float* __restrict__ bnv,
    float* __restrict__ out, int M)
{
  const int lane = threadIdx.x & 31;
  const int wave = threadIdx.x >> 5;   // 0..7
  const int grp  = lane >> 4;          // 0/1 lane half
  const int lidx = lane & 15;
  const int row0 = blockIdx.x * 128 + wave * 16;
  const int r    = row0 + lidx;            // row this lane feeds into A
  const int rc   = (r < M) ? r : (M - 1);  // clamp: OOB rows compute garbage
                                           // in D rows that are never stored

  float scale = 1.0f;
  if (COMB == 1) scale = 1.0f + aux[0];
  float inv = 1.0f;
  if (COMB == 2) inv = 1.0f / fmaxf(aux[rc], 1.0f);

  const float* arow  = A + (size_t)rc * HDIM;
  const float* arow2 = (COMB == 1) ? (A2 + (size_t)rc * HDIM) : A;

  v8f acc[8];
#pragma unroll
  for (int j = 0; j < 8; ++j) acc[j] = (v8f){0.f,0.f,0.f,0.f,0.f,0.f,0.f,0.f};

#pragma unroll
  for (int kk = 0; kk < 4; ++kk) {
    // ---- A fragment: lane holds row rc, K = {base0..base0+7, +16..+23} ----
    const int base0 = kk * 32 + grp * 8;
    v4f qa[4];
    {
      const v4f* p0 = (const v4f*)(arow + base0);
      const v4f* p1 = (const v4f*)(arow + base0 + 16);
      qa[0] = p0[0]; qa[1] = p0[1]; qa[2] = p1[0]; qa[3] = p1[1];
    }
    v16f vf;
#pragma unroll
    for (int t4 = 0; t4 < 4; ++t4)
#pragma unroll
      for (int e4 = 0; e4 < 4; ++e4) vf[t4 * 4 + e4] = qa[t4][e4];

    if (COMB == 1) {
      v4f ga[4];
      const v4f* g0 = (const v4f*)(arow2 + base0);
      const v4f* g1 = (const v4f*)(arow2 + base0 + 16);
      ga[0] = g0[0]; ga[1] = g0[1]; ga[2] = g1[0]; ga[3] = g1[1];
#pragma unroll
      for (int t4 = 0; t4 < 4; ++t4)
#pragma unroll
        for (int e4 = 0; e4 < 4; ++e4)
          vf[t4 * 4 + e4] = scale * vf[t4 * 4 + e4] + ga[t4][e4];
    } else if (COMB == 2) {
#pragma unroll
      for (int e = 0; e < 16; ++e) vf[e] *= inv;
    }

    const v16bf av = __builtin_convertvector(vf, v16bf);

    // ---- 8 col-tiles: B fragment = 32B contiguous per lane (L2 hit) ----
#pragma unroll
    for (int j = 0; j < 8; ++j) {
      union { v4u q[2]; v16bf v; } bf;
      const v4u* wp = (const v4u*)(Wfrag +
          (size_t)((kk * 8 + j) * 32 + lane) * 16);
      bf.q[0] = wp[0];
      bf.q[1] = wp[1];
      acc[j] = __builtin_amdgcn_wmma_f32_16x16x32_bf16(
          false, av, false, bf.v, (short)0, acc[j], false, false);
    }
  }

  // ---- epilogue: bias (+ BN + ReLU), C layout: VGPR rr -> row0+rr+grp*8 ----
#pragma unroll
  for (int j = 0; j < 8; ++j) {
    const int c = j * 16 + lidx;
    const float b0 = bias[c];
    float sc = 0.f, mm = 0.f, bb = 0.f;
    if (BN) {
      sc = bng[c] * rsqrtf(bnv[c] + BN_EPS_F);
      mm = bnm[c];
      bb = bnb[c];
    }
#pragma unroll
    for (int rr = 0; rr < 8; ++rr) {
      const int row = row0 + grp * 8 + rr;
      float y = acc[j][rr] + b0;
      if (BN) { y = (y - mm) * sc + bb; y = fmaxf(y, 0.0f); }
      if (row < M) out[(size_t)row * HDIM + c] = y;
    }
  }
}

// ---------------------------------------------------------------------------
// Edge scatter-add: one wave per edge, lane covers 4 columns (b128 row read,
// 4 f32 atomics). h/agg are 51MB each -> both L2-resident on 192MB L2.
// ---------------------------------------------------------------------------
__global__ __launch_bounds__(256) void gin_scatter(
    const float* __restrict__ h, const int* __restrict__ src,
    const int* __restrict__ dst, float* __restrict__ agg, int E)
{
  const int e    = (int)((blockIdx.x * 256u + threadIdx.x) >> 5);
  const int lane = threadIdx.x & 31;
  if (e >= E) return;
  const int s = src[e];
  const int d = dst[e];
  v4f val = ((const v4f*)(h + (size_t)s * HDIM))[lane];
  float* ap = agg + (size_t)d * HDIM + lane * 4;
  unsafeAtomicAdd(ap + 0, val[0]);
  unsafeAtomicAdd(ap + 1, val[1]);
  unsafeAtomicAdd(ap + 2, val[2]);
  unsafeAtomicAdd(ap + 3, val[3]);
}

// Global mean pool accumulation (batch is sorted, G=512 => hot L2 lines).
__global__ __launch_bounds__(256) void gin_pool(
    const float* __restrict__ h, const int* __restrict__ batch,
    float* __restrict__ pooled, float* __restrict__ counts, int N)
{
  const int n    = (int)((blockIdx.x * 256u + threadIdx.x) >> 5);
  const int lane = threadIdx.x & 31;
  if (n >= N) return;
  const int b = batch[n];
  v4f val = ((const v4f*)(h + (size_t)n * HDIM))[lane];
  float* pp = pooled + (size_t)b * HDIM + lane * 4;
  unsafeAtomicAdd(pp + 0, val[0]);
  unsafeAtomicAdd(pp + 1, val[1]);
  unsafeAtomicAdd(pp + 2, val[2]);
  unsafeAtomicAdd(pp + 3, val[3]);
  if (lane == 0) unsafeAtomicAdd(counts + b, 1.0f);
}

__global__ void zero_f32(float* __restrict__ p, size_t n) {
  size_t i = (size_t)blockIdx.x * blockDim.x + threadIdx.x;
  const size_t stride = (size_t)gridDim.x * blockDim.x;
  for (; i < n; i += stride) p[i] = 0.0f;
}

// ---------------------------------------------------------------------------
extern "C" void kernel_launch(void* const* d_in, const int* in_sizes, int n_in,
                              void* d_out, int out_size, void* d_ws,
                              size_t ws_size, hipStream_t stream) {
  (void)in_sizes; (void)n_in; (void)out_size; (void)ws_size;

  const float* x      = (const float*)d_in[0];
  const int*   eidx   = (const int*)d_in[1];
  const int*   batch  = (const int*)d_in[2];
  const float* encW   = (const float*)d_in[3];
  const float* encB   = (const float*)d_in[4];
  const float* eps    = (const float*)d_in[5];
  const float* m1W    = (const float*)d_in[6];
  const float* m1B    = (const float*)d_in[7];
  const float* m1G    = (const float*)d_in[8];
  const float* m1Beta = (const float*)d_in[9];
  const float* m1M    = (const float*)d_in[10];
  const float* m1V    = (const float*)d_in[11];
  const float* m2W    = (const float*)d_in[12];
  const float* m2B    = (const float*)d_in[13];
  const float* bnG    = (const float*)d_in[14];
  const float* bnB    = (const float*)d_in[15];
  const float* bnM    = (const float*)d_in[16];
  const float* bnV    = (const float*)d_in[17];
  const float* hW1    = (const float*)d_in[18];
  const float* hB1    = (const float*)d_in[19];
  const float* hG     = (const float*)d_in[20];
  const float* hBeta  = (const float*)d_in[21];
  const float* hM     = (const float*)d_in[22];
  const float* hV     = (const float*)d_in[23];
  const float* hW2    = (const float*)d_in[24];
  const float* hB2    = (const float*)d_in[25];

  const int N = NNODES, E = NEDGES, G = NGRAPH;
  const int* src = eidx;
  const int* dst = eidx + E;

  // workspace carve (256B aligned): h, agg, t (51.2MB each), Wfrag, pooled
  char* ws = (char*)d_ws;
  auto carve = [&](size_t bytes) {
    void* p = (void*)ws;
    ws += (bytes + 255) & ~((size_t)255);
    return p;
  };
  float* h    = (float*)carve((size_t)N * HDIM * sizeof(float));
  float* agg  = (float*)carve((size_t)N * HDIM * sizeof(float));
  float* tbuf = (float*)carve((size_t)N * HDIM * sizeof(float));
  unsigned short* Wfrag =
      (unsigned short*)carve((size_t)9 * 16384 * sizeof(unsigned short));
  float* pooled = (float*)carve((size_t)G * HDIM * sizeof(float));
  float* counts = (float*)carve((size_t)G * sizeof(float));

  cvt_weights<<<9, 1024, 0, stream>>>(encW, m1W, m2W, hW1, hW2, Wfrag);

  const int gbN = (N + 127) / 128;  // 782 blocks
  const int gbG = (G + 127) / 128;  // 4 blocks

  // node encoder: h = x @ encW + encB
  gemm128<0, 0><<<gbN, 256, 0, stream>>>(x, nullptr, nullptr, Wfrag, encB,
                                         nullptr, nullptr, nullptr, nullptr,
                                         h, N);

  for (int i = 0; i < 3; ++i) {
    zero_f32<<<1024, 256, 0, stream>>>(agg, (size_t)N * HDIM);
    gin_scatter<<<(E + 7) / 8, 256, 0, stream>>>(h, src, dst, agg, E);
    // t = relu(bn1(((1+eps)*h + agg) @ W1 + b1))
    gemm128<1, 1><<<gbN, 256, 0, stream>>>(
        h, agg, eps + i, Wfrag + (size_t)(1 + i) * 16384, m1B + i * HDIM,
        m1G + i * HDIM, m1Beta + i * HDIM, m1M + i * HDIM, m1V + i * HDIM,
        tbuf, N);
    // h = relu(bn(t @ W2 + b2))
    gemm128<0, 1><<<gbN, 256, 0, stream>>>(
        tbuf, nullptr, nullptr, Wfrag + (size_t)(4 + i) * 16384, m2B + i * HDIM,
        bnG + i * HDIM, bnB + i * HDIM, bnM + i * HDIM, bnV + i * HDIM, h, N);
  }

  zero_f32<<<64, 256, 0, stream>>>(pooled, (size_t)G * HDIM);
  zero_f32<<<1, 256, 0, stream>>>(counts, (size_t)G);
  gin_pool<<<(N + 7) / 8, 256, 0, stream>>>(h, batch, pooled, counts, N);

  // head: t = relu(bn(mean_pool @ hW1 + hB1)); out = t @ hW2 + hB2
  gemm128<2, 1><<<gbG, 256, 0, stream>>>(pooled, nullptr, counts,
                                         Wfrag + (size_t)7 * 16384, hB1, hG,
                                         hBeta, hM, hV, tbuf, G);
  gemm128<0, 0><<<gbG, 256, 0, stream>>>(tbuf, nullptr, nullptr,
                                         Wfrag + (size_t)8 * 16384, hB2,
                                         nullptr, nullptr, nullptr, nullptr,
                                         (float*)d_out, G);
}
